// FNO2d_1305670058366
// MI455X (gfx1250) — matrix operand order repaired and checked by reading, not compiled, not run on hardware
//
#include <hip/hip_runtime.h>
#include <hip/hip_bf16.h>
#include <math.h>

// ---------------------------------------------------------------------------
// FNO2d forward for MI455X (gfx1250, wave32, WMMA + TDM).
// Truncated-DFT reformulation: W-direction DFTs run on VALU, H-direction DFTs become dense GEMMs on
// v_wmma_f32_16x16x32_f16 (f16 inputs, f32 accumulate). The inverse-H complex
// GEMM stages its B matrices into LDS once per workgroup via the Tensor Data
// Mover (tensor_load_to_lds + s_wait_tensorcnt).
// ---------------------------------------------------------------------------

typedef _Float16 f16;
typedef _Float16 v16h __attribute__((ext_vector_type(16)));
typedef _Float16 v8h  __attribute__((ext_vector_type(8)));
typedef float    v8f  __attribute__((ext_vector_type(8)));
typedef unsigned int u32x4 __attribute__((ext_vector_type(4)));
typedef int          i32x4 __attribute__((ext_vector_type(4)));
typedef int          i32x8 __attribute__((ext_vector_type(8)));

#define MODES 12
#define WIDTH 32
#define BATCH 16
#define HDIM  256
#define WDIM  256
#define NPIX  (BATCH * HDIM * WDIM)   // 1,048,576 pixels
#define R24   24                      // kept H-modes (12 low + 12 high)
#define R32   32                      // padded to WMMA tile multiple
#define NSP   (MODES * WIDTH)         // 384 = spectral row length (kw*32+c)

#ifndef __has_builtin
#define __has_builtin(x) 0
#endif

__device__ __forceinline__ float gelu_f(float v) {
  // tanh-approx gelu (matches jax.nn.gelu default)
  return 0.5f * v * (1.0f + tanhf(0.7978845608028654f * (v + 0.044715f * v * v * v)));
}

// ---------------------------------------------------------------------------
// Table generation: truncated DFT matrices (f16, zero-padded to 32 modes),
// W-direction cos/sin tables (f32), inverse-W reconstruction tables, lw->f16.
// ---------------------------------------------------------------------------
__global__ void k_tables(const float* __restrict__ lw,   // [4,32,32]
                         f16* __restrict__ Fh_r, f16* __restrict__ Fh_i, // [32][256]
                         f16* __restrict__ Gr,   f16* __restrict__ Gi,   // [256][32]
                         float* __restrict__ cwf, float* __restrict__ swf, // [12][256]
                         float* __restrict__ cwi, float* __restrict__ swi, // [256][12]
                         f16* __restrict__ lw16) // [4][32][32]
{
  int t = blockIdx.x * blockDim.x + threadIdx.x;   // 0..8191
  const float STEP = 6.283185307179586f / 256.0f;
  if (t < R32 * 256) {  // forward H-DFT rows (e^{-i th}), rows 24..31 zero pad
    int r = t / 256, n = t % 256;
    f16 cr = (f16)0.f, ci = (f16)0.f;
    if (r < R24) {
      int kh = (r < MODES) ? r : (r + 232);        // 12..23 -> 244..255
      float th = (float)((kh * n) & 255) * STEP;
      cr = (f16)cosf(th); ci = (f16)(-sinf(th));
    }
    Fh_r[t] = cr; Fh_i[t] = ci;
  }
  if (t < 256 * R32) {  // inverse H-DFT (e^{+i th}/256), cols 24..31 zero pad
    int n = t / R32, r = t % R32;
    f16 cr = (f16)0.f, ci = (f16)0.f;
    if (r < R24) {
      int kh = (r < MODES) ? r : (r + 232);
      float th = (float)((kh * n) & 255) * STEP;
      cr = (f16)(cosf(th) * (1.0f / 256.0f));
      ci = (f16)(sinf(th) * (1.0f / 256.0f));
    }
    Gr[t] = cr; Gi[t] = ci;
  }
  if (t < MODES * 256) {  // forward W-DFT tables
    int kw = t / 256, m = t % 256;
    float th = (float)((kw * m) & 255) * STEP;
    cwf[t] = cosf(th); swf[t] = sinf(th);
  }
  if (t < 256 * MODES) {  // inverse-W (irfft) reconstruction tables
    int m = t / MODES, kw = t % MODES;
    if (kw == 0) { cwi[t] = 1.0f / 256.0f; swi[t] = 0.0f; }
    else {
      float th = (float)((kw * m) & 255) * STEP;
      cwi[t] = 2.0f * cosf(th) / 256.0f;
      swi[t] = -2.0f * sinf(th) / 256.0f;
    }
  }
  if (t < 4 * WIDTH * WIDTH) lw16[t] = (f16)lw[t];
}

// ---------------------------------------------------------------------------
// fc0: [NPIX,3] x [3,32] + bias -> h16 (f16). K=3, memory bound: plain VALU.
// ---------------------------------------------------------------------------
__global__ void k_fc0(const float* __restrict__ x, const float* __restrict__ w,
                      const float* __restrict__ bias, f16* __restrict__ h16)
{
  int p = blockIdx.x * blockDim.x + threadIdx.x;
  if (p >= NPIX) return;
  float x0 = x[p * 3 + 0], x1 = x[p * 3 + 1], x2 = x[p * 3 + 2];
  #pragma unroll 8
  for (int c = 0; c < WIDTH; ++c) {
    float v = bias[c] + x0 * w[c] + x1 * w[WIDTH + c] + x2 * w[2 * WIDTH + c];
    h16[(long)p * WIDTH + c] = (f16)v;
  }
}

// ---------------------------------------------------------------------------
// Generic batched WMMA GEMM: C[M,N] (+)= alpha * A[M,K] * B[K,N]
// A,B f16 row-major; C f32. One wave per 16x16 C tile; K stepped by 32.
// blockIdx.x -> M tile, blockIdx.y -> N tile, blockIdx.z -> batch.
// ---------------------------------------------------------------------------
__global__ void __launch_bounds__(32)
k_gemm(const f16* __restrict__ A, long sAb, int lda,
       const f16* __restrict__ B, long sBb, int ldb,
       float* __restrict__ C, long sCb, int ldc,
       int K, float alpha, int acc)
{
  __shared__ f16 bt[16][32];
  const int lane = threadIdx.x;
  const int m0 = blockIdx.x * 16;
  const int n0 = blockIdx.y * 16;
  A += (long)blockIdx.z * sAb;
  B += (long)blockIdx.z * sBb;
  C += (long)blockIdx.z * sCb;

  const int idx16 = lane & 15;
  const int half  = lane >> 4;
  v8f accv = {};
  for (int k0 = 0; k0 < K; k0 += 32) {
    // A fragment: two contiguous 8-half chunks from row-major A
    const f16* ap = A + (long)(m0 + idx16) * lda + k0 + half * 8;
    v8h a_lo = *(const v8h*)ap;
    v8h a_hi = *(const v8h*)(ap + 16);
    // Stage B tile transposed: lane loads row (k0+lane), 16 contiguous halves
    const f16* brow = B + (long)(k0 + lane) * ldb + n0;
    v8h b0 = *(const v8h*)brow;
    v8h b1 = *(const v8h*)(brow + 8);
    if (k0 + 32 < K) {  // prefetch next K-step (emits global_prefetch_b8)
      __builtin_prefetch(brow + 32 * (long)ldb, 0, 1);
      __builtin_prefetch(ap + 32, 0, 1);
    }
    __syncthreads();
    #pragma unroll
    for (int j = 0; j < 8; ++j) { bt[j][lane] = b0[j]; bt[j + 8][lane] = b1[j]; }
    __syncthreads();
    const f16* bp = &bt[idx16][half * 8];
    v8h bl = *(const v8h*)bp;
    v8h bh = *(const v8h*)(bp + 16);
    v16h av, bv;
    #pragma unroll
    for (int j = 0; j < 8; ++j) {
      av[j] = a_lo[j]; av[j + 8] = a_hi[j];
      bv[j] = bl[j];   bv[j + 8] = bh[j];
    }
    accv = __builtin_amdgcn_wmma_f32_16x16x32_f16(
        false, av, false, bv, (short)0, accv, false, false);
  }
  const int mb = m0 + half * 8;
  #pragma unroll
  for (int v = 0; v < 8; ++v) {
    float* cp = C + (long)(mb + v) * ldc + n0 + idx16;
    float val = alpha * accv[v];
    if (acc) val += *cp;
    *cp = val;
  }
}

// ---------------------------------------------------------------------------
// Tensor Data Mover: 1-D tile (nelem f16, contiguous) global -> LDS.
// D# built per CDNA5 ISA (group0: count/lds/global/type,
// group1: data_size=2B, tensor_dim0=tile_dim0=nelem, dim1=1).
// ---------------------------------------------------------------------------
__device__ __forceinline__ void tdm_load_1d(const f16* gsrc, f16* lds_dst,
                                            unsigned nelem)
{
#if __has_builtin(__builtin_amdgcn_tensor_load_to_lds)
  unsigned long long ga = (unsigned long long)(uintptr_t)gsrc;
  unsigned lds = (unsigned)(uintptr_t)lds_dst;
  u32x4 g0 = { 1u,                                   // count=1 (valid user D#)
               lds,                                  // lds_addr
               (unsigned)ga,                         // global_addr[31:0]
               (unsigned)((ga >> 32) & 0x01FFFFFFu) | (2u << 30) }; // type=2
  i32x8 g1 = { (int)(1u << 16),                      // data_size=1 -> 2B/elem
               (int)((nelem & 0xFFFFu) << 16),       // tensor_dim0[15:0]
               (int)((nelem >> 16) & 0xFFFFu) | (1 << 16), // dim0 hi, dim1=1
               (int)((nelem & 0xFFFFu) << 16),       // tile_dim0 (<=65535)
               1,                                    // tile_dim1=1
               (int)nelem,                           // tensor_dim0_stride lo32
               0, 0 };
  i32x4 g2 = { 0, 0, 0, 0 };
  i32x4 g3 = { 0, 0, 0, 0 };
#if __clang_major__ >= 23
  i32x8 g4 = { 0, 0, 0, 0, 0, 0, 0, 0 };
  __builtin_amdgcn_tensor_load_to_lds(g0, g1, g2, g3, g4, 0);
#else
  __builtin_amdgcn_tensor_load_to_lds(g0, g1, g2, g3, 0);
#endif
#else
  // Fallback: cooperative vector copy by the calling wave (32 lanes)
  const int lane = threadIdx.x & 31;
  for (unsigned i = lane * 8; i < nelem; i += 32 * 8)
    *(v8h*)(lds_dst + i) = *(const v8h*)(gsrc + i);
#endif
}

// ---------------------------------------------------------------------------
// Fused inverse-H complex GEMM:  Pr = Gr*Zr - Gi*Zi ; Pi = Gr*Zi + Gi*Zr
// One workgroup (8 waves) per batch. B matrices Z2r/Z2i [32,384] f16 staged
// into LDS once via TDM; each wave covers 48 of the 384 16x16 tiles with
// 4 WMMAs per tile. M=256, N=384, K=32 (padded from 24).
// ---------------------------------------------------------------------------
__global__ void __launch_bounds__(256)
k_ihdft(const f16* __restrict__ Gr, const f16* __restrict__ Gi, // [256][32]
        const f16* __restrict__ Z2r, const f16* __restrict__ Z2i, // [B][32][384]
        float* __restrict__ Pr, float* __restrict__ Pi)           // [B][256][384]
{
  __shared__ f16 sBr[R32 * NSP];   // 24 KB
  __shared__ f16 sBi[R32 * NSP];   // 24 KB
  const int b = blockIdx.x;
  const f16* zr = Z2r + (long)b * R32 * NSP;
  const f16* zi = Z2i + (long)b * R32 * NSP;
  float* pr = Pr + (long)b * 256 * NSP;
  float* pi = Pi + (long)b * 256 * NSP;

  if (threadIdx.x < 32) {          // wave 0 issues both TDM loads
    tdm_load_1d(zr, sBr, R32 * NSP);
    tdm_load_1d(zi, sBi, R32 * NSP);
#if __has_builtin(__builtin_amdgcn_s_wait_tensorcnt)
    __builtin_amdgcn_s_wait_tensorcnt(0);
#endif
  }
  __syncthreads();

  const int lane  = threadIdx.x & 31;
  const int wave  = threadIdx.x >> 5;
  const int idx16 = lane & 15;
  const int half  = lane >> 4;

  for (int i = 0; i < 48; ++i) {           // 8 waves x 48 = 384 tiles
    int t  = wave * 48 + i;
    int m0 = (t & 15) * 16;                // 16 M-tiles (rows 0..255)
    int n0 = (t >> 4) * 16;                // 24 N-tiles (cols 0..383)
    // A fragments from global (L2-resident 16KB matrices)
    const f16* arp = Gr + (m0 + idx16) * R32 + half * 8;
    const f16* aip = Gi + (m0 + idx16) * R32 + half * 8;
    v8h ar0 = *(const v8h*)arp, ar1 = *(const v8h*)(arp + 16);
    v8h ai0 = *(const v8h*)aip, ai1 = *(const v8h*)(aip + 16);
    v16h avr, avi, bvr, bvi;
    #pragma unroll
    for (int j = 0; j < 8; ++j) {
      avr[j] = ar0[j]; avr[j + 8] = ar1[j];
      avi[j] = ai0[j]; avi[j + 8] = ai1[j];
    }
    // B fragments from LDS (row-major [k][n], transposed gather per lane)
    #pragma unroll
    for (int e = 0; e < 16; ++e) {
      int kk = (e < 8) ? (half * 8 + e) : (16 + half * 8 + (e - 8));
      bvr[e] = sBr[kk * NSP + n0 + idx16];
      bvi[e] = sBi[kk * NSP + n0 + idx16];
    }
    v8f c1 = {}, c2 = {}, cpi = {};
    c1  = __builtin_amdgcn_wmma_f32_16x16x32_f16(false, avr, false, bvr, (short)0, c1,  false, false);
    c2  = __builtin_amdgcn_wmma_f32_16x16x32_f16(false, avi, false, bvi, (short)0, c2,  false, false);
    cpi = __builtin_amdgcn_wmma_f32_16x16x32_f16(false, avr, false, bvi, (short)0, cpi, false, false);
    cpi = __builtin_amdgcn_wmma_f32_16x16x32_f16(false, avi, false, bvr, (short)0, cpi, false, false);
    const int mb = m0 + half * 8;
    #pragma unroll
    for (int v = 0; v < 8; ++v) {
      pr[(long)(mb + v) * NSP + n0 + idx16] = c1[v] - c2[v];
      pi[(long)(mb + v) * NSP + n0 + idx16] = cpi[v];
    }
  }
}

// ---------------------------------------------------------------------------
// Forward W-DFT: Z[b,r,kw,c] = sum_m Y[b,r,m,c] * e^{-i 2pi kw m/256}.
// ---------------------------------------------------------------------------
__global__ void k_wdft(const float* __restrict__ Yr, const float* __restrict__ Yi,
                       const float* __restrict__ cwf, const float* __restrict__ swf,
                       float* __restrict__ Zr, float* __restrict__ Zi)
{
  int t = blockIdx.x * blockDim.x + threadIdx.x;  // 16*24*12*32
  int c  = t & 31;
  int kw = (t >> 5) % MODES;
  int r  = ((t >> 5) / MODES) % R24;
  int b  = t / (32 * MODES * R24);
  const float* yr = Yr + ((long)(b * R32 + r) * 256) * 32 + c;
  const float* yi = Yi + ((long)(b * R32 + r) * 256) * 32 + c;
  const float* cs = cwf + kw * 256;
  const float* sn = swf + kw * 256;
  float zr = 0.f, zi = 0.f;
  for (int m = 0; m < 256; ++m) {
    float a = yr[m * 32], bb = yi[m * 32], cv = cs[m], sv = sn[m];
    zr += a * cv + bb * sv;
    zi += bb * cv - a * sv;
  }
  long o = ((long)((b * R24 + r) * MODES + kw)) * 32 + c;
  Zr[o] = zr; Zi[o] = zi;
}

// ---------------------------------------------------------------------------
// Spectral mode mixing: out[b,x,y,o] = sum_i Z[b,x,y,i] * W[i,o,x,y] (complex)
// rows 0..11 -> w1 (kh=0..11); rows 12..23 -> w2 (kh=-12..-1).
// Output f16 padded to 32 rows (rows 24..31 zero) for the WMMA GEMM.
// ---------------------------------------------------------------------------
__global__ void k_specmul(const float* __restrict__ Zr, const float* __restrict__ Zi,
                          const float* __restrict__ w1r, const float* __restrict__ w1i,
                          const float* __restrict__ w2r, const float* __restrict__ w2i,
                          f16* __restrict__ Z2r, f16* __restrict__ Z2i)
{
  int t = blockIdx.x * blockDim.x + threadIdx.x;  // 16*32*12*32
  int o  = t & 31;
  int kw = (t >> 5) % MODES;
  int r  = ((t >> 5) / MODES) % R32;
  int b  = t / (32 * MODES * R32);
  long oidx = ((long)((b * R32 + r) * MODES + kw)) * 32 + o;
  if (r >= R24) { Z2r[oidx] = (f16)0.f; Z2i[oidx] = (f16)0.f; return; }
  int x = (r < MODES) ? r : (r - MODES);
  const float* wr = (r < MODES) ? w1r : w2r;
  const float* wi = (r < MODES) ? w1i : w2i;
  const float* zr = Zr + ((long)((b * R24 + r) * MODES + kw)) * 32;
  const float* zi = Zi + ((long)((b * R24 + r) * MODES + kw)) * 32;
  float ar = 0.f, ai = 0.f;
  for (int i = 0; i < WIDTH; ++i) {
    float wrv = wr[((i * WIDTH + o) * MODES + x) * MODES + kw];
    float wiv = wi[((i * WIDTH + o) * MODES + x) * MODES + kw];
    float a = zr[i], bb = zi[i];
    ar += a * wrv - bb * wiv;
    ai += a * wiv + bb * wrv;
  }
  Z2r[oidx] = (f16)ar; Z2i[oidx] = (f16)ai;
}

// ---------------------------------------------------------------------------
// Combine: inverse-W reconstruction (irfft over 12 modes) + pointwise linear
// + bias (+ gelu) -> next h16.
// ---------------------------------------------------------------------------
__global__ void k_combine(const float* __restrict__ Pr, const float* __restrict__ Pi,
                          const float* __restrict__ lin,
                          const float* __restrict__ cwi, const float* __restrict__ swi,
                          const float* __restrict__ lb, int dogelu,
                          f16* __restrict__ h16)
{
  int t = blockIdx.x * blockDim.x + threadIdx.x;  // NPIX*32 = 33.5M
  int c = t & 31;
  int m = (t >> 5) & 255;
  int n = (t >> 13) & 255;
  int b = t >> 21;
  const float* pr = Pr + (long)(b * 256 + n) * NSP + c;
  const float* pi = Pi + (long)(b * 256 + n) * NSP + c;
  const float* cw = cwi + m * MODES;
  const float* sw = swi + m * MODES;
  float s = 0.f;
  #pragma unroll
  for (int kw = 0; kw < MODES; ++kw)
    s += pr[kw * 32] * cw[kw] + pi[kw * 32] * sw[kw];
  float v = s + lin[t] + lb[c];
  if (dogelu) v = gelu_f(v);
  h16[t] = (f16)v;
}

// ---------------------------------------------------------------------------
// Head: per pixel fc1(32->128) + gelu + fc2(128->1), weights cached in LDS.
// ---------------------------------------------------------------------------
__global__ void __launch_bounds__(256)
k_head(const f16* __restrict__ h16,
       const float* __restrict__ fc1_w, const float* __restrict__ fc1_b,
       const float* __restrict__ fc2_w, const float* __restrict__ fc2_b,
       float* __restrict__ out)
{
  __shared__ float w1s[32 * 128];
  __shared__ float b1s[128];
  __shared__ float w2s[128];
  for (int i = threadIdx.x; i < 32 * 128; i += 256) w1s[i] = fc1_w[i];
  if (threadIdx.x < 128) { b1s[threadIdx.x] = fc1_b[threadIdx.x]; w2s[threadIdx.x] = fc2_w[threadIdx.x]; }
  __syncthreads();
  int p = blockIdx.x * blockDim.x + threadIdx.x;
  const f16* hp = h16 + (long)p * 32;
  float hv[32];
  #pragma unroll
  for (int i = 0; i < 32; ++i) hv[i] = (float)hp[i];
  float acc = 0.f;
  for (int j = 0; j < 128; ++j) {
    float tj = b1s[j];
    #pragma unroll
    for (int i = 0; i < 32; ++i) tj += hv[i] * w1s[i * 128 + j];
    acc += gelu_f(tj) * w2s[j];
  }
  out[p] = acc + fc2_b[0];
}

// ---------------------------------------------------------------------------
// Host-side orchestration (graph-capture safe: kernel launches only).
// ---------------------------------------------------------------------------
extern "C" void kernel_launch(void* const* d_in, const int* in_sizes, int n_in,
                              void* d_out, int out_size, void* d_ws, size_t ws_size,
                              hipStream_t stream)
{
  const float* x     = (const float*)d_in[0];
  const float* fc0_w = (const float*)d_in[1];
  const float* fc0_b = (const float*)d_in[2];
  const float* sw1_r = (const float*)d_in[3];
  const float* sw1_i = (const float*)d_in[4];
  const float* sw2_r = (const float*)d_in[5];
  const float* sw2_i = (const float*)d_in[6];
  const float* lw    = (const float*)d_in[7];
  const float* lb    = (const float*)d_in[8];
  const float* fc1_w = (const float*)d_in[9];
  const float* fc1_b = (const float*)d_in[10];
  const float* fc2_w = (const float*)d_in[11];
  const float* fc2_b = (const float*)d_in[12];
  float* out = (float*)d_out;

  // workspace carve-up (256B aligned)
  char* wp = (char*)d_ws;
  auto take = [&](size_t bytes) -> char* {
    char* p = wp; wp += (bytes + 255) & ~(size_t)255; return p;
  };
  f16*   h16   = (f16*)  take((size_t)NPIX * 32 * 2);            // 64 MB
  float* lin   = (float*)take((size_t)NPIX * 32 * 4);            // 128 MB
  float* Yr    = (float*)take((size_t)16 * R32 * 256 * 32 * 4);  // 16 MB
  float* Yi    = (float*)take((size_t)16 * R32 * 256 * 32 * 4);  // 16 MB
  float* Zr    = (float*)take((size_t)16 * R24 * MODES * 32 * 4);
  float* Zi    = (float*)take((size_t)16 * R24 * MODES * 32 * 4);
  f16*   Z2r   = (f16*)  take((size_t)16 * R32 * MODES * 32 * 2);
  f16*   Z2i   = (f16*)  take((size_t)16 * R32 * MODES * 32 * 2);
  float* Pr    = (float*)take((size_t)16 * 256 * NSP * 4);
  float* Pi    = (float*)take((size_t)16 * 256 * NSP * 4);
  f16*   Fh_r  = (f16*)  take((size_t)R32 * 256 * 2);
  f16*   Fh_i  = (f16*)  take((size_t)R32 * 256 * 2);
  f16*   Gr    = (f16*)  take((size_t)256 * R32 * 2);
  f16*   Gi    = (f16*)  take((size_t)256 * R32 * 2);
  float* cwf   = (float*)take((size_t)MODES * 256 * 4);
  float* swf   = (float*)take((size_t)MODES * 256 * 4);
  float* cwi   = (float*)take((size_t)256 * MODES * 4);
  float* swi   = (float*)take((size_t)256 * MODES * 4);
  f16*   lw16  = (f16*)  take((size_t)4 * WIDTH * WIDTH * 2);

  k_tables<<<32, 256, 0, stream>>>(lw, Fh_r, Fh_i, Gr, Gi, cwf, swf, cwi, swi, lw16);
  k_fc0<<<NPIX / 256, 256, 0, stream>>>(x, fc0_w, fc0_b, h16);

  const long WSPEC = (long)WIDTH * WIDTH * MODES * MODES;  // per-block weight stride
  for (int k = 0; k < 4; ++k) {
    // --- forward H-DFT (GEMM): Y[b] = Fh * h16[b]   M=32,N=8192,K=256
    dim3 g1(2, 512, 16);
    k_gemm<<<g1, 32, 0, stream>>>(Fh_r, 0L, 256, h16, (long)256 * 8192, 8192,
                                  Yr, (long)R32 * 8192, 8192, 256, 1.0f, 0);
    k_gemm<<<g1, 32, 0, stream>>>(Fh_i, 0L, 256, h16, (long)256 * 8192, 8192,
                                  Yi, (long)R32 * 8192, 8192, 256, 1.0f, 0);
    // --- forward W-DFT (small, VALU)
    k_wdft<<<(16 * R24 * MODES * 32) / 256, 256, 0, stream>>>(Yr, Yi, cwf, swf, Zr, Zi);
    // --- spectral mode mixing (complex 32x32 per mode, VALU) -> f16 padded
    k_specmul<<<(16 * R32 * MODES * 32) / 256, 256, 0, stream>>>(
        Zr, Zi, sw1_r + k * WSPEC, sw1_i + k * WSPEC,
        sw2_r + k * WSPEC, sw2_i + k * WSPEC, Z2r, Z2i);
    // --- inverse H-DFT: fused complex WMMA GEMM, TDM-staged B, 1 WG/batch
    k_ihdft<<<16, 256, 0, stream>>>(Gr, Gi, Z2r, Z2i, Pr, Pi);
    // --- pointwise linear (GEMM): lin = h16 @ lw[k]   M=NPIX,N=32,K=32
    dim3 g5(NPIX / 16, 2, 1);
    k_gemm<<<g5, 32, 0, stream>>>(h16, 0L, 32, lw16 + k * WIDTH * WIDTH, 0L, 32,
                                  lin, 0L, 32, 32, 1.0f, 0);
    // --- inverse-W irfft + linear + bias (+gelu) -> next h16
    k_combine<<<(NPIX * 32) / 256, 256, 0, stream>>>(
        Pr, Pi, lin, cwi, swi, lb + k * WIDTH, (k < 3) ? 1 : 0, h16);
  }

  // head: fc1 + gelu + fc2, fused (avoids 512MB intermediate)
  k_head<<<NPIX / 256, 256, 0, stream>>>(h16, fc1_w, fc1_b, fc2_w, fc2_b, out);
  (void)in_sizes; (void)n_in; (void)out_size; (void)ws_size;
}
